// ConvUnit_17308718203063
// MI455X (gfx1250) — matrix operand order repaired and checked by reading, not compiled, not run on hardware
//
#include <hip/hip_runtime.h>

// ---------------------------------------------------------------------------
// Bit-sliced conv2d for MI455X (gfx1250, wave32, WMMA, async-to-LDS staging).
// y = sum_b clamp(conv(bit_b(int8(x)), w), -1024, 1023) * scale_b + bias
// Implicit GEMM, operands arranged as D = W x Bits:
//   A = weights (16 ch x 32 K, fp16), B = bit-planes (32 K x 16 pixels, fp16
//   {0,1}), D/C = f32 16x16 with lane = pixel -> coalesced output stores.
// Block = 256 threads = 8 waves; wave = 32ch x 16pix, 8 bit-plane accums per
// 16x16 tile (128 VGPRs). Pixel tile (32 rows x 576 K int8 im2col) staged in
// LDS via global_load_async_to_lds_b8; K-loop reads it with 1 ds_load_b128.
// ---------------------------------------------------------------------------

typedef __attribute__((ext_vector_type(16))) _Float16 v16h;
typedef __attribute__((ext_vector_type(8)))  float    v8f;

#define N_BATCH 64
#define C_IN    64
#define H_IN    56
#define W_IN    56
#define K_OUT   128
#define H_OUT   54
#define W_OUT   54
#define K_TOT   (C_IN * 9)                  // 576
#define M_TOT   (N_BATCH * H_OUT * W_OUT)   // 186624 = 5832 * 32 exactly
#define PIX_OUT (H_OUT * W_OUT)             // 2916
#define XPIX    (C_IN * H_IN * W_IN)        // 200704 per image
#define XPLANE  (H_IN * W_IN)               // 3136 per channel
#define M_TILE  32
#define LDS_STRIDE 592                      // 576 + 16 pad (multiple of 16)

// --- prep 1: fp32 x -> clamped int8 (clip(-128,127).astype(int8)) ----------
__global__ void quantize_x_kernel(const float* __restrict__ x,
                                  signed char* __restrict__ xq, int n) {
  int i = blockIdx.x * blockDim.x + threadIdx.x;
  int stride = gridDim.x * blockDim.x;
  for (; i < n; i += stride) {
    float v = x[i];
    v = fminf(fmaxf(v, -128.0f), 127.0f);
    xq[i] = (signed char)(int)v;            // trunc toward zero = astype(int8)
  }
}

// --- prep 2: fp32 OIHW weights -> fp16 Wt[128][576] (already row-major) ----
__global__ void convert_w_kernel(const float* __restrict__ w,
                                 _Float16* __restrict__ wt, int n) {
  int i = blockIdx.x * blockDim.x + threadIdx.x;
  if (i < n) wt[i] = (_Float16)w[i];
}

// --- main WMMA kernel ------------------------------------------------------
__global__ __launch_bounds__(256)
void bitconv_wmma_kernel(const signed char* __restrict__ xq,
                         const _Float16*   __restrict__ wt,
                         const float*      __restrict__ bias,
                         float*            __restrict__ out) {
  // Single static LDS allocation -> starts at LDS offset 0 (used by async ops).
  __shared__ alignas(16) signed char s_a[M_TILE * LDS_STRIDE];  // 18944 B

  const int tid = threadIdx.x;
  const int m0  = blockIdx.x * M_TILE;

  // ---- stage im2col int8 pixel tile into LDS via async per-lane copies ----
  {
    const int row  = tid >> 3;                  // 32 rows, 8 threads per row
    const int mrow = m0 + row;
    const int ni   = mrow / PIX_OUT;
    const int pr   = mrow % PIX_OUT;
    const long pixbase = (long)ni * XPIX + (pr / W_OUT) * W_IN + (pr % W_OUT);
    const signed char* xrow = xq + pixbase;
#pragma unroll 4
    for (int k = (tid & 7); k < K_TOT; k += 8) {
      const int c = k / 9, q = k - 9 * c;
      const int koff = c * XPLANE + (q / 3) * W_IN + (q % 3);
      unsigned lds_off = (unsigned)(row * LDS_STRIDE + k);
      unsigned long long gaddr = (unsigned long long)(xrow + koff);
      asm volatile("global_load_async_to_lds_b8 %0, %1, off"
                   :: "v"(lds_off), "v"(gaddr) : "memory");
    }
  }
  asm volatile("s_wait_asynccnt 0" ::: "memory");
  __syncthreads();

  const int lane = tid & 31;                 // wave32
  const int wave = tid >> 5;
  const int mh   = wave >> 2;                // which 16-pixel half of the tile
  const int nq   = wave & 3;                 // which 32-channel quad
  const int hi   = (lane >> 4) & 1;          // lane half selects K sub-blocks
  const int l15  = lane & 15;

  // B side (bits): lane = pixel column; lanes<16 hold K=k0..k0+15 contiguous, same order as A.
  const unsigned abase = (unsigned)((mh * 16 + l15) * LDS_STRIDE);
  // A side (weights): lane = channel row; K = kb..kb+15 contiguous.
  const _Float16* wrow0 = wt + (long)(nq * 32 + l15) * K_TOT;
  const _Float16* wrow1 = wrow0 + 16 * (long)K_TOT;

  v8f acc[8][2] = {};                        // [bit][channel-tile], 128 VGPRs

  struct alignas(16) H8 { _Float16 h[8]; };

#pragma unroll 2
  for (int k0 = 0; k0 < K_TOT; k0 += 32) {
    // ---- A fragments: weights, two contiguous 16B chunks per 16-ch tile ----
    const int kb = k0 + (hi ? 16 : 0);
    union WF { v16h v; H8 p[2]; } wf0, wf1;
    wf0.p[0] = *(const H8*)(wrow0 + kb);
    wf0.p[1] = *(const H8*)(wrow0 + kb + 8);
    wf1.p[0] = *(const H8*)(wrow1 + kb);
    wf1.p[1] = *(const H8*)(wrow1 + kb + 8);

    // ---- B source: 16 contiguous im2col bytes -> one ds_load_b128 ----
    const int kbB = k0 + (hi ? 16 : 0);
    const unsigned* ap =
        (const unsigned*)__builtin_assume_aligned(&s_a[abase + kbB], 16);
    const unsigned d0 = ap[0], d1 = ap[1], d2 = ap[2], d3 = ap[3];

    // P[j] = {byte(2j), 0, byte(2j+1), 0} : one v_perm_b32 each
    unsigned P[8];
    P[0] = __builtin_amdgcn_perm(d0, d0, 0x0c010c00u);
    P[1] = __builtin_amdgcn_perm(d0, d0, 0x0c030c02u);
    P[2] = __builtin_amdgcn_perm(d1, d1, 0x0c010c00u);
    P[3] = __builtin_amdgcn_perm(d1, d1, 0x0c030c02u);
    P[4] = __builtin_amdgcn_perm(d2, d2, 0x0c010c00u);
    P[5] = __builtin_amdgcn_perm(d2, d2, 0x0c030c02u);
    P[6] = __builtin_amdgcn_perm(d3, d3, 0x0c010c00u);
    P[7] = __builtin_amdgcn_perm(d3, d3, 0x0c030c02u);

    // 16 WMMAs per K-step: 8 bit-planes x 2 channel tiles, shared expansion.
    // ((P>>b)&0x00010001)*0x3C00 -> packed fp16 {0,1} pair, no carry.
#pragma unroll
    for (int b = 0; b < 8; ++b) {
      union { v16h v; unsigned u[8]; } xf;
#pragma unroll
      for (int j = 0; j < 8; ++j)
        xf.u[j] = ((P[j] >> b) & 0x00010001u) * 0x3C00u;
      acc[b][0] = __builtin_amdgcn_wmma_f32_16x16x32_f16(
          false, wf0.v, false, xf.v, (short)0, acc[b][0], false, false);
      acc[b][1] = __builtin_amdgcn_wmma_f32_16x16x32_f16(
          false, wf1.v, false, xf.v, (short)0, acc[b][1], false, false);
    }
  }

  // ---- epilogue: lane = pixel -> coalesced stores down each channel ----
  const float scales[8] = {1.f, 2.f, 4.f, 8.f, 16.f, 32.f, 64.f, -128.f};
  const int mo = m0 + mh * 16 + l15;         // this lane's output pixel
  const int ni = mo / PIX_OUT;
  const int pr = mo % PIX_OUT;
  float* obase = out + (long)ni * (K_OUT * PIX_OUT) + pr;
#pragma unroll
  for (int t = 0; t < 2; ++t) {
#pragma unroll
    for (int e = 0; e < 8; ++e) {            // C/D: VGPR e -> channel row e(+8)
      const int ch = nq * 32 + t * 16 + e + hi * 8;
      float y = bias[ch];
#pragma unroll
      for (int b = 0; b < 8; ++b) {
        float v = fminf(fmaxf(acc[b][t][e], -1024.0f), 1023.0f);
        y += v * scales[b];
      }
      obase[(long)ch * PIX_OUT] = y;         // lanes span 16 consecutive ow
    }
  }
}

extern "C" void kernel_launch(void* const* d_in, const int* in_sizes, int n_in,
                              void* d_out, int out_size, void* d_ws, size_t ws_size,
                              hipStream_t stream) {
  (void)in_sizes; (void)n_in; (void)out_size; (void)ws_size;
  const float* x    = (const float*)d_in[0];
  const float* w    = (const float*)d_in[1];
  const float* bias = (const float*)d_in[2];
  float* out = (float*)d_out;

  // Workspace: [int8 xq : 12,845,056 B][fp16 Wt : 147,456 B] (16B aligned)
  signed char* xq = (signed char*)d_ws;
  _Float16*    wt = (_Float16*)((char*)d_ws + (size_t)N_BATCH * C_IN * H_IN * W_IN);

  const int nx = N_BATCH * C_IN * H_IN * W_IN;   // 12,845,056
  quantize_x_kernel<<<2048, 256, 0, stream>>>(x, xq, nx);

  const int nw = K_OUT * K_TOT;                  // 73,728
  convert_w_kernel<<<(nw + 255) / 256, 256, 0, stream>>>(w, wt, nw);

  bitconv_wmma_kernel<<<M_TOT / M_TILE, 256, 0, stream>>>(xq, wt, bias, out);
}
